// LSTMAutoencoder_9809705304320
// MI455X (gfx1250) — matrix operand order repaired and checked by reading, not compiled, not run on hardware
//
#include <hip/hip_runtime.h>

// LSTM autoencoder: B=256, T=512, F=64, H=128, 2+2 layers.
// Strategy: parallel WMMA GEMMs for input projections (xg) and final projection;
// latency-critical recurrence done by 16 persistent workgroups (16 batch rows each),
// W_hh held entirely in registers as WMMA B-fragments (128 VGPRs/wave), h/gates
// exchanged through LDS with two barriers per timestep. xg for the current step is
// loaded into registers at the top of the iteration so the global loads overlap the
// WMMA phase instead of serializing after the barrier.

#define BB   256
#define TT   512
#define FF   64
#define HH   128
#define G4H  512   // 4*H

typedef __attribute__((ext_vector_type(16))) _Float16 v16h;
typedef __attribute__((ext_vector_type(8)))  _Float16 v8h;
typedef __attribute__((ext_vector_type(4)))  _Float16 v4h;
typedef __attribute__((ext_vector_type(8)))  float    v8f;
typedef __attribute__((ext_vector_type(4)))  float    v4f;

// Load 8 consecutive f32 and convert to 8 f16.
__device__ __forceinline__ v8h cvt8(const float* p) {
  v4f a = *(const v4f*)p;
  v4f b = *(const v4f*)(p + 4);
  v4h ah = __builtin_convertvector(a, v4h);
  v4h bh = __builtin_convertvector(b, v4h);
  v8h r;
  ((v4h*)&r)[0] = ah;
  ((v4h*)&r)[1] = bh;
  return r;
}
__device__ __forceinline__ v8h ld8(const float* p)    { return cvt8(p); }
__device__ __forceinline__ v8h ld8(const _Float16* p) { return *(const v8h*)p; }

__device__ __forceinline__ float sigm(float x) {
  return __builtin_amdgcn_rcpf(1.0f + __expf(-x));
}
__device__ __forceinline__ float tanh_(float x) {
  float e = __expf(-2.0f * x);
  return (1.0f - e) * __builtin_amdgcn_rcpf(1.0f + e);
}

__device__ __forceinline__ v8f wmma_f16(v16h a, v16h b, v8f c) {
  // (neg_a, A, neg_b, B, c_mod, C, reuse_a, reuse_b)
  return __builtin_amdgcn_wmma_f32_16x16x32_f16(false, a, false, b, (short)0, c,
                                                false, false);
}

// -------------------------------------------------------------------------
// xg GEMM: XG[row, 0:512] = In[row, 0:KI] @ Wih^T + (bih + bhh), f16 output.
// Block = 256 thr = 8 waves; each wave owns 64 output cols (4 N-tiles) of one
// 16-row tile. Row address = rbase+lm for both lane halves (A layout).
// -------------------------------------------------------------------------
template <typename TIN>
__global__ __launch_bounds__(256)
void xg_gemm_kernel(const TIN* __restrict__ In, long long row_stride, long long row_off,
                    const float* __restrict__ Wih,
                    const float* __restrict__ bih, const float* __restrict__ bhh,
                    _Float16* __restrict__ XG, int KI)
{
  const int wave = threadIdx.x >> 5;
  const int lane = threadIdx.x & 31;
  const int half = lane >> 4;
  const int lm   = lane & 15;
  const long long rbase = (long long)blockIdx.x * 16;
  const int nb = wave * 64;

  v8f D[4];
#pragma unroll
  for (int nt = 0; nt < 4; ++nt)
#pragma unroll
    for (int i = 0; i < 8; ++i) D[nt][i] = 0.0f;

  const int nkc = KI >> 5;                 // K chunks of 32
  const TIN* arow = In + (rbase + lm) * row_stride + row_off;
#pragma unroll
  for (int kc = 0; kc < 4; ++kc) {
    if (kc >= nkc) break;
    // A fragment: elems 0..7 <- K kb+8*half, elems 8..15 <- K kb+16+8*half
    v16h A;
    const TIN* ap = arow + kc * 32 + half * 8;
    ((v8h*)&A)[0] = ld8(ap);
    ((v8h*)&A)[1] = ld8(ap + 16);
#pragma unroll
    for (int nt = 0; nt < 4; ++nt) {
      // B fragment: column n = W row (nb+nt*16+lm); elems 0..7 <- K kb+8*half, elems 8..15 <- K kb+16+8*half
      const float* p = Wih + (long long)(nb + nt * 16 + lm) * KI + kc * 32 + half * 8;
      v16h Bv;
      ((v8h*)&Bv)[0] = cvt8(p);
      ((v8h*)&Bv)[1] = cvt8(p + 16);
      D[nt] = wmma_f16(A, Bv, D[nt]);
    }
  }
  // D layout: vgpr r, lane -> (m = r + 8*half, n = lm)
#pragma unroll
  for (int nt = 0; nt < 4; ++nt) {
    int col = nb + nt * 16 + lm;
    float bias = bih[col] + bhh[col];
    _Float16* op = XG + (rbase + half * 8) * G4H + col;
#pragma unroll
    for (int r = 0; r < 8; ++r)
      op[(long long)r * G4H] = (_Float16)(D[nt][r] + bias);
  }
}

// -------------------------------------------------------------------------
// Recurrent kernel: one block per 16 batch rows, full T loop, no global sync.
// 8 waves; wave w owns gate cols [w*64, w*64+64). W_hh B-fragments live in
// registers for the whole loop. Per step: load xg(t) early, 16 WMMAs/wave ->
// gates to LDS -> barrier -> fp32 gate math (c in registers) -> new h to LDS
// + global -> barrier.
// -------------------------------------------------------------------------
__global__ __launch_bounds__(256)
void lstm_rec_kernel(const _Float16* __restrict__ xg, long long xg_bstride,
                     long long xg_tstride,
                     const float* __restrict__ Whh,
                     _Float16* __restrict__ hs_out)
{
  __shared__ __align__(16) _Float16 hsh[16 * HH];   // current h, f16
  __shared__ __align__(16) float    gsh[16 * G4H];  // pre-activation gates, f32

  const int wave = threadIdx.x >> 5;
  const int lane = threadIdx.x & 31;
  const int half = lane >> 4;
  const int lm   = lane & 15;
  const int bbase = blockIdx.x * 16;
  const int nb = wave * 64;
  (void)wave;

  // Preload W_hh as B fragments (constant across timesteps). 16 x v16h = 128 VGPRs.
  v16h Bf[4][4];
#pragma unroll
  for (int nt = 0; nt < 4; ++nt) {
    const float* wr = Whh + (long long)(nb + nt * 16 + lm) * HH;
#pragma unroll
    for (int kc = 0; kc < 4; ++kc) {
      const float* p = wr + kc * 32 + half * 8;
      v16h Bv;
      ((v8h*)&Bv)[0] = cvt8(p);
      ((v8h*)&Bv)[1] = cvt8(p + 16);
      Bf[nt][kc] = Bv;
    }
  }

  // h0 = 0
  {
    v8h z = {};
    *(v8h*)(hsh + threadIdx.x * 8) = z;
  }
  float c[8];
#pragma unroll
  for (int e = 0; e < 8; ++e) c[e] = 0.0f;

  // Elementwise ownership: thread -> (batch row em, hidden cols ej..ej+7), fixed
  // across timesteps so the cell state c stays in registers.
  const int em = threadIdx.x >> 4;
  const int ej = (threadIdx.x & 15) * 8;
  const _Float16* xg_b = xg + (long long)(bbase + em) * xg_bstride + ej;
  _Float16* out_b = hs_out + (long long)(bbase + em) * TT * HH + ej;

  __syncthreads();

#pragma unroll 1
  for (int t = 0; t < TT; ++t) {
    // ---- issue this step's gate-input loads FIRST so they overlap the WMMAs ----
    const _Float16* xgp = xg_b + t * xg_tstride;
    v8h xi  = *(const v8h*)(xgp + 0 * HH);
    v8h xf  = *(const v8h*)(xgp + 1 * HH);
    v8h xgg = *(const v8h*)(xgp + 2 * HH);
    v8h xo  = *(const v8h*)(xgp + 3 * HH);
    __builtin_prefetch(xgp + xg_tstride, 0, 3);  // next step, into near caches

    // ---- GEMM phase: gates_pre = h @ Whh^T (16 x 512), reads hsh ----
    v8f D[4];
#pragma unroll
    for (int nt = 0; nt < 4; ++nt)
#pragma unroll
      for (int i = 0; i < 8; ++i) D[nt][i] = 0.0f;

#pragma unroll
    for (int kc = 0; kc < 4; ++kc) {
      v16h A;
      const _Float16* hp = hsh + lm * HH + kc * 32 + half * 8;
      ((v8h*)&A)[0] = *(const v8h*)hp;
      ((v8h*)&A)[1] = *(const v8h*)(hp + 16);
#pragma unroll
      for (int nt = 0; nt < 4; ++nt)
        D[nt] = wmma_f16(A, Bf[nt][kc], D[nt]);
    }
#pragma unroll
    for (int nt = 0; nt < 4; ++nt) {
      float* g = gsh + (half * 8) * G4H + nb + nt * 16 + lm;
#pragma unroll
      for (int r = 0; r < 8; ++r) g[r * G4H] = D[nt][r];
    }
    __syncthreads();  // gsh visible; all hsh reads of this step are done

    // ---- elementwise phase: gate order i, f, g, o ----
    const float* gr = gsh + em * G4H + ej;
    v8h hv;
#pragma unroll
    for (int e = 0; e < 8; ++e) {
      float gi = gr[0 * HH + e] + (float)xi[e];
      float gf = gr[1 * HH + e] + (float)xf[e];
      float gg = gr[2 * HH + e] + (float)xgg[e];
      float go = gr[3 * HH + e] + (float)xo[e];
      float cc = sigm(gf) * c[e] + sigm(gi) * tanh_(gg);
      c[e] = cc;
      hv[e] = (_Float16)(sigm(go) * tanh_(cc));
    }
    *(v8h*)(hsh + em * HH + ej) = hv;          // new h for next step
    *(v8h*)(out_b + (long long)t * HH) = hv;   // h sequence to global (f16)
    __syncthreads();  // hsh visible; gsh reads done before next GEMM overwrites
  }
}

// -------------------------------------------------------------------------
// Final projection: out[row, f] = hs[row, :] . W_out[f, :] + b_out[f], f32 out.
// One 16x16 tile per wave, K = 128.
// -------------------------------------------------------------------------
__global__ __launch_bounds__(256)
void proj_kernel(const _Float16* __restrict__ hs, const float* __restrict__ Wout,
                 const float* __restrict__ bout, float* __restrict__ out)
{
  const int wave = threadIdx.x >> 5;
  const int lane = threadIdx.x & 31;
  const int half = lane >> 4;
  const int lm   = lane & 15;
  const int tile = blockIdx.x * 8 + wave;   // (B*T/16) * (F/16) tiles
  const int rt = tile >> 2;
  const int ct = tile & 3;
  const long long rbase = (long long)rt * 16;

  v8f D;
#pragma unroll
  for (int i = 0; i < 8; ++i) D[i] = 0.0f;

#pragma unroll
  for (int kc = 0; kc < 4; ++kc) {
    v16h A;
    const _Float16* ap = hs + (rbase + lm) * HH + kc * 32 + half * 8;
    ((v8h*)&A)[0] = *(const v8h*)ap;
    ((v8h*)&A)[1] = *(const v8h*)(ap + 16);
    const float* p = Wout + (long long)(ct * 16 + lm) * HH + kc * 32 + half * 8;
    v16h Bv;
    ((v8h*)&Bv)[0] = cvt8(p);
    ((v8h*)&Bv)[1] = cvt8(p + 16);
    D = wmma_f16(A, Bv, D);
  }
  const int col = ct * 16 + lm;
  const float bias = bout[col];
  float* op = out + (rbase + half * 8) * FF + col;
#pragma unroll
  for (int r = 0; r < 8; ++r) op[(long long)r * FF] = D[r] + bias;
}

// -------------------------------------------------------------------------
extern "C" void kernel_launch(void* const* d_in, const int* in_sizes, int n_in,
                              void* d_out, int out_size, void* d_ws, size_t ws_size,
                              hipStream_t stream)
{
  (void)in_sizes; (void)n_in; (void)out_size; (void)ws_size;

  const float* x      = (const float*)d_in[0];
  const float* e0_Wih = (const float*)d_in[1];
  const float* e0_Whh = (const float*)d_in[2];
  const float* e0_bih = (const float*)d_in[3];
  const float* e0_bhh = (const float*)d_in[4];
  const float* e1_Wih = (const float*)d_in[5];
  const float* e1_Whh = (const float*)d_in[6];
  const float* e1_bih = (const float*)d_in[7];
  const float* e1_bhh = (const float*)d_in[8];
  const float* d0_Wih = (const float*)d_in[9];
  const float* d0_Whh = (const float*)d_in[10];
  const float* d0_bih = (const float*)d_in[11];
  const float* d0_bhh = (const float*)d_in[12];
  const float* d1_Wih = (const float*)d_in[13];
  const float* d1_Whh = (const float*)d_in[14];
  const float* d1_bih = (const float*)d_in[15];
  const float* d1_bhh = (const float*)d_in[16];
  const float* W_out  = (const float*)d_in[17];
  const float* b_out  = (const float*)d_in[18];

  // Workspace layout (f16): XG [B*T, 4H] = 128 MB, two h-sequence buffers 32 MB each.
  const size_t XG_BYTES = (size_t)BB * TT * G4H * sizeof(_Float16);
  const size_t HS_BYTES = (size_t)BB * TT * HH * sizeof(_Float16);
  char* ws = (char*)d_ws;
  _Float16* XG  = (_Float16*)ws;
  _Float16* hsA = (_Float16*)(ws + XG_BYTES);
  _Float16* hsB = (_Float16*)(ws + XG_BYTES + HS_BYTES);

  dim3 blk(256);
  const int RT = (BB * TT) / 16;  // 8192 row tiles

  // Encoder layer 0 (input F=64)
  xg_gemm_kernel<float><<<RT, blk, 0, stream>>>(x, FF, 0, e0_Wih, e0_bih, e0_bhh, XG, FF);
  lstm_rec_kernel<<<BB / 16, blk, 0, stream>>>(XG, (long long)TT * G4H, G4H, e0_Whh, hsA);
  // Encoder layer 1
  xg_gemm_kernel<_Float16><<<RT, blk, 0, stream>>>(hsA, HH, 0, e1_Wih, e1_bih, e1_bhh, XG, HH);
  lstm_rec_kernel<<<BB / 16, blk, 0, stream>>>(XG, (long long)TT * G4H, G4H, e1_Whh, hsB);
  // Decoder layer 0: input is enc final hidden (row t=T-1 of hsB), constant over t,
  // so xg is a single [B, 4H] GEMM and the recurrence uses xg_tstride = 0.
  xg_gemm_kernel<_Float16><<<BB / 16, blk, 0, stream>>>(hsB, (long long)TT * HH,
                                                        (long long)(TT - 1) * HH,
                                                        d0_Wih, d0_bih, d0_bhh, XG, HH);
  lstm_rec_kernel<<<BB / 16, blk, 0, stream>>>(XG, G4H, 0, d0_Whh, hsA);
  // Decoder layer 1
  xg_gemm_kernel<_Float16><<<RT, blk, 0, stream>>>(hsA, HH, 0, d1_Wih, d1_bih, d1_bhh, XG, HH);
  lstm_rec_kernel<<<BB / 16, blk, 0, stream>>>(XG, (long long)TT * G4H, G4H, d1_Whh, hsB);
  // Output projection
  proj_kernel<<<(RT * 4) / 8, blk, 0, stream>>>(hsB, W_out, b_out, (float*)d_out);
}
